// HyperSETLoRAAttention_71614284694167
// MI455X (gfx1250) — matrix operand order repaired and checked by reading, compiled, not run on hardware
//
#include <hip/hip_runtime.h>
#include <hip/hip_bf16.h>
#include <math.h>

// ---------------------------------------------------------------------------
// HyperSET LoRA attention for gfx1250 (MI455X), bf16 WMMA pipeline.
//   K0: W = qkv+delta -> Wbf (row-major), WTbf (transposed); x -> bf16
//   K1: w = x@W^T via wmma, per-(token,head) rms -> Q=wn*g1*g2/8, K=wn, Vt=wn*g3 (transposed)
//   K2: att stats: exact column (max,sum) + per-strip row partials
//   K2b: reduce row partials -> exact row (max,sum)
//   K3: recompute att tiles, P = exp/row + exp/col, P@V via LDS transpose -> y bf16
//   K4: out = y @ W via wmma (fp32 out)
// ---------------------------------------------------------------------------

typedef __bf16 bf16_t;
typedef bf16_t v16bf __attribute__((ext_vector_type(16)));
typedef bf16_t v8bf  __attribute__((ext_vector_type(8)));
typedef float  v8f   __attribute__((ext_vector_type(8)));

#define DEV __device__ __forceinline__

static constexpr int   Bn  = 4;
static constexpr int   Tn  = 2048;
static constexpr int   Cn  = 512;
static constexpr int   Hn  = 8;
static constexpr int   Dn  = 64;
static constexpr int   BH  = Bn * Hn;
static constexpr float EPSF = 1.1920929e-07f;

// ---- WMMA operand loaders (CDNA5 ISA 7.12.2 layouts) ----------------------
// A (16x32 bf16, MxK): lanes 0-15 row=lane, k={k0..k0+7, k0+16..k0+23};
//                      lanes 16-31 row=lane-16, k shifted by +8.
DEV v16bf load_a_tile(const bf16_t* __restrict__ base, int ld, int m0, int k0, int lane) {
  int m  = m0 + (lane & 15);
  int kb = k0 + ((lane >> 4) << 3);
  const bf16_t* p = base + (size_t)m * ld + kb;
  v8bf lo = *(const v8bf*)(p);
  v8bf hi = *(const v8bf*)(p + 16);
  return __builtin_shufflevector(lo, hi, 0,1,2,3,4,5,6,7,8,9,10,11,12,13,14,15);
}

// B (32x16 bf16, KxN) from a transposed (NxK row-major) source: lane n=lane&15
// holds k={kb..kb+7, kb+16..kb+23}, kb = k0 + 8*(lane>>4) (same order as A).
DEV v16bf load_b_tile(const bf16_t* __restrict__ baseT, int ld, int n0, int k0, int lane) {
  int n  = n0 + (lane & 15);
  int kb = k0 + ((lane >> 4) << 3);
  const bf16_t* p = baseT + (size_t)n * ld + kb;
  v8bf lo = *(const v8bf*)(p);
  v8bf hi = *(const v8bf*)(p + 16);
  return __builtin_shufflevector(lo, hi, 0,1,2,3,4,5,6,7,8,9,10,11,12,13,14,15);
}

DEV v8f wmma_bf16(v16bf a, v16bf b, v8f c) {
  return __builtin_amdgcn_wmma_f32_16x16x32_bf16(false, a, false, b, (short)0, c, false, false);
}

DEV v8f v8f_zero() { v8f z = {0.f,0.f,0.f,0.f,0.f,0.f,0.f,0.f}; return z; }

// ---------------------------------------------------------------------------
// K0: prep  W = qkv_w + delta_w ;  bf16 conversions
// ---------------------------------------------------------------------------
__global__ void k_prep(const float* __restrict__ x,
                       const float* __restrict__ dw,
                       const float* __restrict__ qw,
                       bf16_t* __restrict__ xbf,
                       bf16_t* __restrict__ Wbf,
                       bf16_t* __restrict__ WTbf) {
  int idx = blockIdx.x * blockDim.x + threadIdx.x;
  if (idx < Cn * Cn) {
    int r = idx / Cn, c = idx % Cn;
    float w = qw[idx] + dw[idx];            // W[r][c]
    Wbf [(size_t)r * Cn + c] = (bf16_t)w;   // Bt for GEMM1 (x @ W^T)
    WTbf[(size_t)c * Cn + r] = (bf16_t)w;   // Bt for GEMM2 (y @ W)
  }
  if (idx < Bn * Tn * Cn) xbf[idx] = (bf16_t)x[idx];
}

// ---------------------------------------------------------------------------
// K1: w = x @ W^T (64x64 tile per WG, one head wide), rms, emit Q/K/Vt bf16
// ---------------------------------------------------------------------------
__global__ void __launch_bounds__(128)
k_gemm1_qkv(const bf16_t* __restrict__ xbf, const bf16_t* __restrict__ Wbf,
            const float* __restrict__ g1, const float* __restrict__ g2,
            const float* __restrict__ g3,
            bf16_t* __restrict__ Qb, bf16_t* __restrict__ Kb,
            bf16_t* __restrict__ Vtb) {
  int lane = threadIdx.x & 31;
  int wv   = threadIdx.x >> 5;
  int h    = blockIdx.y;
  int m0   = blockIdx.x * 64 + wv * 16;

  v8f acc[4];
  for (int nt = 0; nt < 4; ++nt) acc[nt] = v8f_zero();

  for (int k0 = 0; k0 < Cn; k0 += 32) {
    v16bf a = load_a_tile(xbf, Cn, m0, k0, lane);
    for (int nt = 0; nt < 4; ++nt) {
      v16bf b = load_b_tile(Wbf, Cn, h * 64 + nt * 16, k0, lane);
      acc[nt] = wmma_bf16(a, b, acc[nt]);
    }
  }

  int mb = (lane >> 4) << 3;
  for (int r = 0; r < 8; ++r) {
    float s = 0.f;
    for (int nt = 0; nt < 4; ++nt) s += acc[nt][r] * acc[nt][r];
    for (int off = 1; off < 16; off <<= 1) s += __shfl_xor(s, off, 32);
    float inv = rsqrtf(s * (1.0f / Dn) + EPSF);

    int mg = m0 + mb + r;
    int b  = mg >> 11;              // / Tn
    int t  = mg & (Tn - 1);
    int bh = b * Hn + h;
    for (int nt = 0; nt < 4; ++nt) {
      int   f  = nt * 16 + (lane & 15);
      float wn = acc[nt][r] * inv;
      float gg = g1[f] * g2[f] * 0.125f;   // fold g1*g2 and 1/sqrt(D) into Q
      Qb [((size_t)bh * Tn + t) * Dn + f] = (bf16_t)(wn * gg);
      Kb [((size_t)bh * Tn + t) * Dn + f] = (bf16_t)wn;
      Vtb[((size_t)bh * Dn + f) * Tn + t] = (bf16_t)(wn * g3[f]);
    }
  }
}

// ---------------------------------------------------------------------------
// K2: per (bh, 64-col strip): exact column (max,sum) + per-strip row partials
// ---------------------------------------------------------------------------
__global__ void __launch_bounds__(128)
k_stats(const bf16_t* __restrict__ Qb, const bf16_t* __restrict__ Kb,
        float* __restrict__ colm, float* __restrict__ cols,
        float* __restrict__ rowm_part, float* __restrict__ rows_part) {
  __shared__ float lm[4][16];
  __shared__ float ls[4][16];
  int lane  = threadIdx.x & 31;
  int wv    = threadIdx.x >> 5;
  int bh    = blockIdx.x;
  int strip = blockIdx.y;
  int j0    = strip * 64 + wv * 16;

  const bf16_t* Qh = Qb + (size_t)bh * Tn * Dn;
  const bf16_t* Kh = Kb + (size_t)bh * Tn * Dn;

  v16bf b0 = load_b_tile(Kh, Dn, j0, 0, lane);
  v16bf b1 = load_b_tile(Kh, Dn, j0, 32, lane);

  float cm = -3.0e38f, cs = 0.f;
  int mb = (lane >> 4) << 3;

  for (int i0 = 0; i0 < Tn; i0 += 16) {
    v16bf a0 = load_a_tile(Qh, Dn, i0, 0, lane);
    v16bf a1 = load_a_tile(Qh, Dn, i0, 32, lane);
    v8f c = v8f_zero();
    c = wmma_bf16(a0, b0, c);
    c = wmma_bf16(a1, b1, c);

    // ---- column online stats (lane pair l, l+16 share column) ----
    float tmax = c[0];
    for (int r = 1; r < 8; ++r) tmax = fmaxf(tmax, c[r]);
    tmax = fmaxf(tmax, __shfl_xor(tmax, 16, 32));
    float newm = fmaxf(cm, tmax);
    float tsum = 0.f;
    for (int r = 0; r < 8; ++r) tsum += __expf(c[r] - newm);
    tsum += __shfl_xor(tsum, 16, 32);
    cs = cs * __expf(cm - newm) + tsum;
    cm = newm;

    // ---- row stats over this wave's 16 columns ----
    for (int r = 0; r < 8; ++r) {
      float rm = c[r];
      for (int off = 1; off < 16; off <<= 1) rm = fmaxf(rm, __shfl_xor(rm, off, 32));
      float rs = __expf(c[r] - rm);
      for (int off = 1; off < 16; off <<= 1) rs += __shfl_xor(rs, off, 32);
      if ((lane & 15) == 0) { lm[wv][mb + r] = rm; ls[wv][mb + r] = rs; }
    }
    __syncthreads();
    if (threadIdx.x < 16) {
      float m = lm[0][threadIdx.x];
      for (int w = 1; w < 4; ++w) m = fmaxf(m, lm[w][threadIdx.x]);
      float s = 0.f;
      for (int w = 0; w < 4; ++w) s += ls[w][threadIdx.x] * __expf(lm[w][threadIdx.x] - m);
      size_t ridx = ((size_t)bh * 32 + strip) * Tn + i0 + threadIdx.x;
      rowm_part[ridx] = m;
      rows_part[ridx] = s;
    }
    __syncthreads();
  }

  if (lane < 16) {
    colm[(size_t)bh * Tn + j0 + lane] = cm;
    cols[(size_t)bh * Tn + j0 + lane] = cs;
  }
}

// ---------------------------------------------------------------------------
// K2b: combine 32 row partials -> exact row (max, sum)
// ---------------------------------------------------------------------------
__global__ void k_rowreduce(const float* __restrict__ rowm_part,
                            const float* __restrict__ rows_part,
                            float* __restrict__ rowm, float* __restrict__ rows) {
  int idx = blockIdx.x * blockDim.x + threadIdx.x;
  if (idx >= BH * Tn) return;
  int bh = idx / Tn, i = idx % Tn;
  float m = -3.0e38f;
  for (int s = 0; s < 32; ++s)
    m = fmaxf(m, rowm_part[((size_t)bh * 32 + s) * Tn + i]);
  float sum = 0.f;
  for (int s = 0; s < 32; ++s)
    sum += rows_part[((size_t)bh * 32 + s) * Tn + i] *
           __expf(rowm_part[((size_t)bh * 32 + s) * Tn + i] - m);
  rowm[idx] = m;
  rows[idx] = sum;
}

// ---------------------------------------------------------------------------
// K3: recompute att tiles, P = exp/rowsum + exp/colsum, y = P @ V
// ---------------------------------------------------------------------------
__global__ void __launch_bounds__(128)
k_attn(const bf16_t* __restrict__ Qb, const bf16_t* __restrict__ Kb,
       const bf16_t* __restrict__ Vtb,
       const float* __restrict__ colm, const float* __restrict__ cols,
       const float* __restrict__ rowm, const float* __restrict__ rows,
       bf16_t* __restrict__ ybf) {
  __shared__ float pbuf[4][16][36];   // per-wave 16x32 P staging (+pad)
  int lane = threadIdx.x & 31;
  int wv   = threadIdx.x >> 5;
  int bh   = blockIdx.x;
  int i0   = blockIdx.y * 64 + wv * 16;

  const bf16_t* Qh = Qb  + (size_t)bh * Tn * Dn;
  const bf16_t* Kh = Kb  + (size_t)bh * Tn * Dn;
  const bf16_t* Vh = Vtb + (size_t)bh * Dn * Tn;

  v16bf a0 = load_a_tile(Qh, Dn, i0, 0, lane);
  v16bf a1 = load_a_tile(Qh, Dn, i0, 32, lane);

  int mb = (lane >> 4) << 3;
  float rm8[8], ri8[8];
  for (int r = 0; r < 8; ++r) {
    int i = i0 + mb + r;
    rm8[r] = rowm[(size_t)bh * Tn + i];
    ri8[r] = 1.f / rows[(size_t)bh * Tn + i];
  }

  v8f y[4];
  for (int ft = 0; ft < 4; ++ft) y[ft] = v8f_zero();

  for (int j0 = 0; j0 < Tn; j0 += 32) {
    for (int jt = 0; jt < 2; ++jt) {
      int jj = j0 + jt * 16;
      v16bf b0 = load_b_tile(Kh, Dn, jj, 0, lane);
      v16bf b1 = load_b_tile(Kh, Dn, jj, 32, lane);
      v8f c = v8f_zero();
      c = wmma_bf16(a0, b0, c);
      c = wmma_bf16(a1, b1, c);

      int   n   = jj + (lane & 15);
      float cmv = colm[(size_t)bh * Tn + n];
      float ciw = 1.f / cols[(size_t)bh * Tn + n];
      for (int r = 0; r < 8; ++r) {
        float p = __expf(c[r] - rm8[r]) * ri8[r] + __expf(c[r] - cmv) * ciw;
        pbuf[wv][mb + r][jt * 16 + (lane & 15)] = p;
      }
    }
    asm volatile("s_wait_dscnt 0" ::: "memory");  // LDS RAW within wave

    // re-read P in WMMA A-operand layout (on-chip transpose via LDS)
    v16bf a2;
    {
      int m  = lane & 15;
      int kb = (lane >> 4) << 3;
      for (int e = 0; e < 8; ++e) {
        a2[e]     = (bf16_t)pbuf[wv][m][kb + e];
        a2[e + 8] = (bf16_t)pbuf[wv][m][kb + 16 + e];
      }
    }
    for (int ft = 0; ft < 4; ++ft) {
      v16bf bv = load_b_tile(Vh, Tn, ft * 16, j0, lane);  // Vt is (D x T) = Bt layout
      y[ft] = wmma_bf16(a2, bv, y[ft]);
    }
  }

  int b = bh >> 3, h = bh & 7;
  for (int ft = 0; ft < 4; ++ft)
    for (int r = 0; r < 8; ++r) {
      int i = i0 + mb + r;
      ybf[((size_t)(b * Tn + i)) * Cn + h * Dn + ft * 16 + (lane & 15)] =
          (bf16_t)y[ft][r];
    }
}

// ---------------------------------------------------------------------------
// K4: out = y @ W   (fp32 output)
// ---------------------------------------------------------------------------
__global__ void __launch_bounds__(128)
k_gemm2(const bf16_t* __restrict__ ybf, const bf16_t* __restrict__ WTbf,
        float* __restrict__ out) {
  int lane = threadIdx.x & 31;
  int wv   = threadIdx.x >> 5;
  int m0   = blockIdx.x * 64 + wv * 16;
  int n0   = blockIdx.y * 64;

  v8f acc[4];
  for (int nt = 0; nt < 4; ++nt) acc[nt] = v8f_zero();

  for (int k0 = 0; k0 < Cn; k0 += 32) {
    v16bf a = load_a_tile(ybf, Cn, m0, k0, lane);
    for (int nt = 0; nt < 4; ++nt) {
      v16bf b = load_b_tile(WTbf, Cn, n0 + nt * 16, k0, lane);
      acc[nt] = wmma_bf16(a, b, acc[nt]);
    }
  }

  int mb = (lane >> 4) << 3;
  for (int nt = 0; nt < 4; ++nt)
    for (int r = 0; r < 8; ++r)
      out[(size_t)(m0 + mb + r) * Cn + n0 + nt * 16 + (lane & 15)] = acc[nt][r];
}

// ---------------------------------------------------------------------------
extern "C" void kernel_launch(void* const* d_in, const int* in_sizes, int n_in,
                              void* d_out, int out_size, void* d_ws, size_t ws_size,
                              hipStream_t stream) {
  (void)in_sizes; (void)n_in; (void)out_size; (void)ws_size;

  const float* x  = (const float*)d_in[0];
  const float* dw = (const float*)d_in[1];
  const float* qw = (const float*)d_in[2];
  const float* g1 = (const float*)d_in[3];
  const float* g2 = (const float*)d_in[4];
  const float* g3 = (const float*)d_in[5];
  float* out = (float*)d_out;

  char* ws = (char*)d_ws;
  size_t off = 0;
  auto alloc = [&](size_t bytes) -> void* {
    void* p = ws + off;
    off = (off + bytes + 255) & ~(size_t)255;
    return p;
  };

  bf16_t* xbf  = (bf16_t*)alloc((size_t)Bn * Tn * Cn * 2);
  bf16_t* Wbf  = (bf16_t*)alloc((size_t)Cn * Cn * 2);
  bf16_t* WTbf = (bf16_t*)alloc((size_t)Cn * Cn * 2);
  bf16_t* Qb   = (bf16_t*)alloc((size_t)BH * Tn * Dn * 2);
  bf16_t* Kb   = (bf16_t*)alloc((size_t)BH * Tn * Dn * 2);
  bf16_t* Vtb  = (bf16_t*)alloc((size_t)BH * Dn * Tn * 2);
  float*  colm = (float*)alloc((size_t)BH * Tn * 4);
  float*  colsum = (float*)alloc((size_t)BH * Tn * 4);
  float*  rowm_part = (float*)alloc((size_t)BH * 32 * Tn * 4);
  float*  rows_part = (float*)alloc((size_t)BH * 32 * Tn * 4);
  float*  rowm = (float*)alloc((size_t)BH * Tn * 4);
  float*  rowsum = (float*)alloc((size_t)BH * Tn * 4);
  bf16_t* ybf  = (bf16_t*)alloc((size_t)Bn * Tn * Cn * 2);

  k_prep<<<(Bn * Tn * Cn + 255) / 256, 256, 0, stream>>>(x, dw, qw, xbf, Wbf, WTbf);

  k_gemm1_qkv<<<dim3(Bn * Tn / 64, Hn), 128, 0, stream>>>(
      xbf, Wbf, g1, g2, g3, Qb, Kb, Vtb);

  k_stats<<<dim3(BH, Tn / 64), 128, 0, stream>>>(
      Qb, Kb, colm, colsum, rowm_part, rows_part);

  k_rowreduce<<<(BH * Tn + 255) / 256, 256, 0, stream>>>(
      rowm_part, rows_part, rowm, rowsum);

  k_attn<<<dim3(BH, Tn / 64), 128, 0, stream>>>(
      Qb, Kb, Vtb, colm, colsum, rowm, rowsum, ybf);

  k_gemm2<<<dim3(Bn * Tn / 64, Cn / 64), 128, 0, stream>>>(ybf, WTbf, out);
}
